// EGNN_encoder_62672162783749
// MI455X (gfx1250) — matrix operand order repaired and checked by reading, not compiled, not run on hardware
//
#include <hip/hip_runtime.h>

// ---------------------------------------------------------------------------
// EGNN encoder for MI455X (gfx1250, wave32, WMMA f16 16x16x32).
// B=64, N=64, F=64, H=128, L=3. Edge-concat GEMM factored into Hi/Hj node
// GEMMs; per-edge MLPs fused into one WMMA kernel streaming over K, with the
// Hj tile double-buffered via global_load_async_to_lds_b128 (ASYNCcnt).
// ---------------------------------------------------------------------------

#define NB 64
#define NN 64
#define NF 64
#define NH 128
#define E2PAD 544   // 514 padded to 17*32
#define HJOFF 544
#define LDHIHJ 1088

typedef __attribute__((ext_vector_type(16))) _Float16 v16h;
typedef __attribute__((ext_vector_type(8)))  _Float16 v8h;
typedef __attribute__((ext_vector_type(8)))  float    v8f;
typedef __attribute__((ext_vector_type(4)))  float    v4f;

__device__ __forceinline__ v8f wmma16(v16h a, v16h b, v8f c) {
  return __builtin_amdgcn_wmma_f32_16x16x32_f16(false, a, false, b, (short)0, c,
                                                false, false);
}
// fast silu/sigmoid: v_exp_f32 + v_rcp_f32 (avoid IEEE div_fixup chains)
__device__ __forceinline__ float silu_(float x) {
  return x * __builtin_amdgcn_rcpf(1.f + __expf(-x));
}
__device__ __forceinline__ float sigm_(float x) {
  return __builtin_amdgcn_rcpf(1.f + __expf(-x));
}
__device__ __forceinline__ v8f v8zero() {
  v8f z = {0.f, 0.f, 0.f, 0.f, 0.f, 0.f, 0.f, 0.f};
  return z;
}

// ---------------------------------------------------------------------------
// Adjacency from initial coordinates: adj[b,i,j] = (d2 < 2.5^2) & mi & mj & i!=j
// ---------------------------------------------------------------------------
__global__ void adj_kernel(const float* __restrict__ pos,
                           const int* __restrict__ mask,
                           float* __restrict__ adj) {
  int b = blockIdx.x >> 6;
  int i = blockIdx.x & 63;
  int j = threadIdx.x;
  const float* pi = pos + (b * NN + i) * 3;
  const float* pj = pos + (b * NN + j) * 3;
  float dx = pi[0] - pj[0], dy = pi[1] - pj[1], dz = pi[2] - pj[2];
  float d2 = dx * dx + dy * dy + dz * dz;
  bool a = (d2 < 6.25f) && (mask[b * NN + i] != 0) && (mask[b * NN + j] != 0) && (i != j);
  adj[(size_t)(b * NN + i) * NN + j] = a ? 1.f : 0.f;
}

// ---------------------------------------------------------------------------
// Pack an f32 weight [K x N] (row stride ldw) into WMMA B-fragment order:
// dst[((kt*NT + nt)*32 + lane)*16 + h] = W[kt*32 + (lane/16)*8 + (h&7) + (h>>3)*16][nt*16 + lane%16]
// zero-padded beyond K / Nv.
// ---------------------------------------------------------------------------
__global__ void pack_kernel(const float* __restrict__ W, int ldw, int K, int Nv,
                            int KT, int NT, _Float16* __restrict__ dst) {
  size_t total = (size_t)KT * NT * 512;
  for (size_t idx = (size_t)blockIdx.x * blockDim.x + threadIdx.x; idx < total;
       idx += (size_t)gridDim.x * blockDim.x) {
    int h = (int)(idx & 15);
    int lanei = (int)((idx >> 4) & 31);
    size_t t = idx >> 9;
    int nt = (int)(t % NT);
    int kt = (int)(t / NT);
    int k = kt * 32 + ((lanei >> 4) << 3) + (h & 7) + ((h >> 3) << 4);
    int n = nt * 16 + (lanei & 15);
    float v = (k < K && n < Nv) ? W[(size_t)k * ldw + n] : 0.f;
    dst[idx] = (_Float16)v;
  }
}

// ---------------------------------------------------------------------------
// Generic WMMA GEMM: C[m,n] = act(A @ Wpacked + bias) (+ Res). One 16x16 output
// tile per wave. A is f32 row-major [16*Mt x 32*KT] (fully K-padded by caller,
// 16B aligned, lda multiple of 4).
// ---------------------------------------------------------------------------
template <int ACT, bool RES>
__global__ __launch_bounds__(256) void gemm16(
    const float* __restrict__ A, int lda, int Mt, int KT,
    const _Float16* __restrict__ Wp, const float* __restrict__ bias, int nbias,
    float* __restrict__ C, int ldc, int NT, const float* __restrict__ Res, int ldr) {
  int wid = blockIdx.x * (blockDim.x >> 5) + (threadIdx.x >> 5);
  if (wid >= Mt * NT) return;
  int mt = wid / NT, nt = wid % NT;
  int lane = threadIdx.x & 31;
  int arow = lane & 15;
  int kgrp = lane >> 4;

  v8f acc = v8zero();
  for (int kt = 0; kt < KT; ++kt) {
    const v4f* Ar4 =
        (const v4f*)(A + (size_t)(mt * 16 + arow) * lda + kt * 32 + (kgrp << 3));
    v4f x0 = Ar4[0], x1 = Ar4[1];  // k .. k+7
    v4f x2 = Ar4[4], x3 = Ar4[5];  // k+16 .. k+23
    v16h a;
#pragma unroll
    for (int h = 0; h < 4; ++h) {
      a[h] = (_Float16)x0[h];
      a[h + 4] = (_Float16)x1[h];
      a[h + 8] = (_Float16)x2[h];
      a[h + 12] = (_Float16)x3[h];
    }
    v16h b = *(const v16h*)(Wp + (((size_t)kt * NT + nt) * 32 + lane) * 16);
    acc = wmma16(a, b, acc);
  }
  int col = nt * 16 + (lane & 15);
  float bv = (bias != nullptr && col < nbias) ? bias[col] : 0.f;
#pragma unroll
  for (int r = 0; r < 8; ++r) {
    float v = acc[r] + bv;
    if (ACT == 1) v = silu_(v);
    int orow = mt * 16 + (kgrp << 3) + r;
    if (RES) v += Res[(size_t)orow * ldr + col];
    C[(size_t)orow * ldc + col] = v;
  }
}

// ---------------------------------------------------------------------------
// Fused per-edge kernel. Block = (batch b, 16-row i tile), 128 threads / 4 waves.
// Each wave handles one j per chunk (16 chunks of 4 j's cover N=64).
// Hj tiles are double-buffered in LDS, filled by async DMA
// (global_load_async_to_lds_b128) overlapped with the previous chunk's WMMAs.
// ---------------------------------------------------------------------------
__global__ __launch_bounds__(128) void edge_kernel(
    const float* __restrict__ HiHj, const float* __restrict__ xin,
    const float* __restrict__ adj, const float* __restrict__ hglob,
    const _Float16* __restrict__ w2p, const float* __restrict__ e_b2,
    const float* __restrict__ g_w, const float* __restrict__ g_b,
    const _Float16* __restrict__ c1p, const float* __restrict__ c_b1,
    const float* __restrict__ c_w2, const float* __restrict__ c_b2,
    const float* __restrict__ e_w1, const float* __restrict__ coors_scale,
    float* __restrict__ xout, float* __restrict__ hcat) {
  __shared__ __align__(16) _Float16 sHi[16][E2PAD];
  __shared__ __align__(16) float sHj[2][4][E2PAD];   // double-buffered DMA dest
  __shared__ __align__(16) float sC[E2PAD];
  __shared__ float sX[NN][3];
  __shared__ float sMi[16][NH];
  __shared__ float sXa[16][4];
  __shared__ __align__(16) _Float16 sM[4][16][NH];

  const int b = blockIdx.x >> 2;
  const int i0 = (blockIdx.x & 3) << 4;
  const int tid = threadIdx.x;
  const int wv = tid >> 5;
  const int lane = tid & 31;
  const int arow = lane & 15;
  const int kgrp = lane >> 4;
  const int colw = lane & 15;

  // async DMA of one 4-row Hj chunk (4 x 544 f32 = 544 b128 transfers)
  auto issue_hj = [&](int jcc, int bufi) {
    const float* gbase = HiHj + (size_t)(b * NN + jcc * 4) * LDHIHJ + HJOFF;
    for (int idx = tid; idx < 544; idx += 128) {
      int r = idx / 136;
      int cq = idx - r * 136;
      const float* g = gbase + (size_t)r * LDHIHJ + (size_t)cq * 4;
      unsigned la = (unsigned)(unsigned long long)(&sHj[bufi][r][cq * 4]);
      asm volatile("global_load_async_to_lds_b128 %0, %1, off"
                   :: "v"(la), "v"(g) : "memory");
    }
  };

  issue_hj(0, 0);  // prologue: overlaps with the sHi/sC/sX cooperative loads

  for (int idx = tid; idx < 16 * E2PAD; idx += 128) {
    int r = idx / E2PAD, k = idx % E2PAD;
    sHi[r][k] = (_Float16)HiHj[(size_t)(b * NN + i0 + r) * LDHIHJ + k];
  }
  for (int idx = tid; idx < E2PAD; idx += 128)
    sC[idx] = (idx < 514) ? e_w1[256 * 514 + idx] : 0.f;
  for (int idx = tid; idx < NN * 3; idx += 128)
    sX[idx / 3][idx % 3] = xin[(size_t)(b * NN) * 3 + idx];
  for (int idx = tid; idx < 16 * NH; idx += 128) sMi[idx >> 7][idx & 127] = 0.f;
  if (tid < 64) sXa[tid >> 2][tid & 3] = 0.f;
  __syncthreads();

  const float cscale = coors_scale[0];
  const float gb0 = g_b[0];
  const float cb2v = c_b2[0];
  const float* adjrow = adj + (size_t)(b * NN + i0) * NN;
  float eb2v[8], gwv[8];
#pragma unroll
  for (int nt = 0; nt < 8; ++nt) {
    eb2v[nt] = e_b2[nt * 16 + colw];
    gwv[nt] = g_w[nt * 16 + colw];
  }

  for (int jc = 0; jc < 16; ++jc) {
    // drain this chunk's DMA, sync, then kick off the next chunk's DMA so it
    // overlaps with all WMMA work below.
    asm volatile("s_wait_asynccnt 0" ::: "memory");
    __syncthreads();
    if (jc + 1 < 16) {
      issue_hj(jc + 1, (jc + 1) & 1);
      if (jc + 2 < 16) {  // prefetch chunk jc+2 into L2 (global_prefetch_b8)
        const float* pf = HiHj + (size_t)(b * NN + (jc + 2) * 4) * LDHIHJ + HJOFF;
        __builtin_prefetch(pf + (size_t)tid * 17, 0, 0);
      }
    }
    const float(*HJ)[E2PAD] = sHj[jc & 1];

    const int j = jc * 4 + wv;
    float dxr = sX[i0 + arow][0] - sX[j][0];
    float dyr = sX[i0 + arow][1] - sX[j][1];
    float dzr = sX[i0 + arow][2] - sX[j][2];
    float d2 = dxr * dxr + dyr * dyr + dzr * dzr;

    // ---- m = silu(e) @ W2 : K = 544, N = 128 -------------------------------
    v8f macc[8];
#pragma unroll
    for (int nt = 0; nt < 8; ++nt) macc[nt] = v8zero();
    for (int kt = 0; kt < 17; ++kt) {
      int kb = kt * 32 + (kgrp << 3);
      // vectorized LDS reads: 2x b128 (f16) + 8x b128 (f32)
      v8h hi0 = *(const v8h*)&sHi[arow][kb];
      v8h hi1 = *(const v8h*)&sHi[arow][kb + 16];
      v4f hj00 = *(const v4f*)&HJ[wv][kb];
      v4f hj01 = *(const v4f*)&HJ[wv][kb + 4];
      v4f hj10 = *(const v4f*)&HJ[wv][kb + 16];
      v4f hj11 = *(const v4f*)&HJ[wv][kb + 20];
      v4f c00 = *(const v4f*)&sC[kb];
      v4f c01 = *(const v4f*)&sC[kb + 4];
      v4f c10 = *(const v4f*)&sC[kb + 16];
      v4f c11 = *(const v4f*)&sC[kb + 20];
      v16h a;
#pragma unroll
      for (int hh = 0; hh < 4; ++hh) {
        float v0 = (float)hi0[hh] + hj00[hh] + d2 * c00[hh];
        float v1 = (float)hi0[hh + 4] + hj01[hh] + d2 * c01[hh];
        float v2 = (float)hi1[hh] + hj10[hh] + d2 * c10[hh];
        float v3 = (float)hi1[hh + 4] + hj11[hh] + d2 * c11[hh];
        a[hh] = (_Float16)silu_(v0);
        a[hh + 4] = (_Float16)silu_(v1);
        a[hh + 8] = (_Float16)silu_(v2);
        a[hh + 12] = (_Float16)silu_(v3);
      }
#pragma unroll
      for (int nt = 0; nt < 8; ++nt) {
        v16h bf = *(const v16h*)(w2p + (((size_t)kt * 8 + nt) * 32 + lane) * 16);
        macc[nt] = wmma16(a, bf, macc[nt]);
      }
    }

    // ---- bias + silu + soft-edge gate --------------------------------------
    float gpart[8] = {0, 0, 0, 0, 0, 0, 0, 0};
#pragma unroll
    for (int nt = 0; nt < 8; ++nt) {
#pragma unroll
      for (int r = 0; r < 8; ++r) {
        float v = silu_(macc[nt][r] + eb2v[nt]);
        macc[nt][r] = v;
        gpart[r] += v * gwv[nt];
      }
    }
#pragma unroll
    for (int r = 0; r < 8; ++r) {
      gpart[r] += __shfl_xor(gpart[r], 1);
      gpart[r] += __shfl_xor(gpart[r], 2);
      gpart[r] += __shfl_xor(gpart[r], 4);
      gpart[r] += __shfl_xor(gpart[r], 8);
      gpart[r] = sigm_(gpart[r] + gb0);
    }
#pragma unroll
    for (int nt = 0; nt < 8; ++nt)
#pragma unroll
      for (int r = 0; r < 8; ++r) macc[nt][r] *= gpart[r];

    // ---- stash gated m (f16) for re-fragmentation; aggregate m_i -----------
#pragma unroll
    for (int nt = 0; nt < 8; ++nt)
#pragma unroll
      for (int r = 0; r < 8; ++r)
        sM[wv][kgrp * 8 + r][nt * 16 + colw] = (_Float16)macc[nt][r];
#pragma unroll
    for (int r = 0; r < 8; ++r) {
      int ir = kgrp * 8 + r;
      if (adjrow[(size_t)ir * NN + j] != 0.f) {
#pragma unroll
        for (int nt = 0; nt < 8; ++nt)
          atomicAdd(&sMi[ir][nt * 16 + colw], macc[nt][r]);
      }
    }
    __syncthreads();

    // ---- coors MLP: w = silu(m @ C1 + cb1) . c2 + cb2 ----------------------
    // hoist the 4 A-fragments of m once (registers), reuse across all 32 nt2.
    v16h a2f[4];
#pragma unroll
    for (int kt2 = 0; kt2 < 4; ++kt2) {
      int kb2 = kt2 * 32 + (kgrp << 3);
      v8h lo = *(const v8h*)&sM[wv][arow][kb2];
      v8h hi = *(const v8h*)&sM[wv][arow][kb2 + 16];
      a2f[kt2] = __builtin_shufflevector(lo, hi, 0, 1, 2, 3, 4, 5, 6, 7, 8, 9,
                                         10, 11, 12, 13, 14, 15);
    }
    float wsum[8] = {0, 0, 0, 0, 0, 0, 0, 0};
    for (int nt2 = 0; nt2 < 32; ++nt2) {
      v8f cacc = v8zero();
#pragma unroll
      for (int kt2 = 0; kt2 < 4; ++kt2) {
        v16h bf = *(const v16h*)(c1p + (((size_t)kt2 * 32 + nt2) * 32 + lane) * 16);
        cacc = wmma16(a2f[kt2], bf, cacc);
      }
      float cb = c_b1[nt2 * 16 + colw];
      float c2v = c_w2[nt2 * 16 + colw];
#pragma unroll
      for (int r = 0; r < 8; ++r) wsum[r] += silu_(cacc[r] + cb) * c2v;
    }
#pragma unroll
    for (int r = 0; r < 8; ++r) {
      wsum[r] += __shfl_xor(wsum[r], 1);
      wsum[r] += __shfl_xor(wsum[r], 2);
      wsum[r] += __shfl_xor(wsum[r], 4);
      wsum[r] += __shfl_xor(wsum[r], 8);
    }
    if ((lane & 15) < 8) {
      int ir = kgrp * 8 + (lane & 7);
      float av = adjrow[(size_t)ir * NN + j];
      if (av != 0.f) {
        float wv_ = wsum[lane & 7] + cb2v;
        float dx = sX[i0 + ir][0] - sX[j][0];
        float dy = sX[i0 + ir][1] - sX[j][1];
        float dz = sX[i0 + ir][2] - sX[j][2];
        float dd = dx * dx + dy * dy + dz * dz;
        float inv = __builtin_amdgcn_rsqf(fmaxf(dd, 1e-16f)) * cscale;
        atomicAdd(&sXa[ir][0], wv_ * dx * inv);
        atomicAdd(&sXa[ir][1], wv_ * dy * inv);
        atomicAdd(&sXa[ir][2], wv_ * dz * inv);
      }
    }
  }
  __syncthreads();

  // ---- epilogue: [h | m_i] -> hcat, x += dx --------------------------------
  for (int idx = tid; idx < 16 * NH; idx += 128) {
    int r = idx >> 7, c = idx & 127;
    size_t row = (size_t)(b * NN + i0 + r);
    hcat[row * 256 + c] = hglob[row * NH + c];
    hcat[row * 256 + NH + c] = sMi[r][c];
  }
  if (tid < 48) {
    int r = tid / 3, c = tid % 3;
    int row = b * NN + i0 + r;
    xout[row * 3 + c] = sX[i0 + r][c] + sXa[r][c];
  }
}

// ---------------------------------------------------------------------------
// Final masked scatter: out = [ h (B*N*128) | x (B*N*3) ]
// ---------------------------------------------------------------------------
__global__ void finalize_kernel(const int* __restrict__ mask,
                                const float* __restrict__ ho,
                                const float* __restrict__ hb,
                                const float* __restrict__ xe,
                                const float* __restrict__ pos,
                                float* __restrict__ out) {
  const int HTOT = NB * NN * NH;
  const int TOT = HTOT + NB * NN * 3;
  for (int idx = blockIdx.x * blockDim.x + threadIdx.x; idx < TOT;
       idx += gridDim.x * blockDim.x) {
    if (idx < HTOT) {
      int row = idx >> 7;
      out[idx] = mask[row] ? ho[idx] : hb[idx];
    } else {
      int k = idx - HTOT;
      int row = k / 3;
      out[idx] = mask[row] ? xe[k] : pos[k];
    }
  }
}

// ---------------------------------------------------------------------------
extern "C" void kernel_launch(void* const* d_in, const int* in_sizes, int n_in,
                              void* d_out, int out_size, void* d_ws, size_t ws_size,
                              hipStream_t stream) {
  (void)in_sizes; (void)n_in; (void)out_size; (void)ws_size;
  const float* atom  = (const float*)d_in[0];
  const float* pos   = (const float*)d_in[1];
  const int*   mask  = (const int*)d_in[2];
  const float* emb_w = (const float*)d_in[3];
  const float* emb_b = (const float*)d_in[4];
  const float* mlp_w = (const float*)d_in[5];
  const float* mlp_b = (const float*)d_in[6];
  const float* how   = (const float*)d_in[7];
  const float* hob   = (const float*)d_in[8];

  char* base = (char*)d_ws;
  size_t off = 0;
  auto carve = [&](size_t bytes) -> void* {
    void* p = base + off;
    off = (off + bytes + 255) & ~(size_t)255;
    return p;
  };
  float* adjw  = (float*)carve((size_t)NB * NN * NN * 4);
  float* hbuf  = (float*)carve((size_t)4096 * NH * 4);
  float* hcat  = (float*)carve((size_t)4096 * 256 * 4);
  float* nmbuf = (float*)carve((size_t)4096 * 256 * 4);
  float* HiHj  = (float*)carve((size_t)4096 * LDHIHJ * 4);
  float* xA    = (float*)carve((size_t)4096 * 3 * 4);
  float* xB    = (float*)carve((size_t)4096 * 3 * 4);
  float* htmp  = (float*)carve((size_t)4096 * NH * 4);
  float* hbase = (float*)carve((size_t)4096 * NH * 4);
  _Float16* w1a_p  = (_Float16*)carve(69632ull * 2);
  _Float16* w1b_p  = (_Float16*)carve(69632ull * 2);
  _Float16* w2_p   = (_Float16*)carve(69632ull * 2);
  _Float16* c1_p   = (_Float16*)carve(65536ull * 2);
  _Float16* n1_p   = (_Float16*)carve(65536ull * 2);
  _Float16* n2_p   = (_Float16*)carve(32768ull * 2);
  _Float16* emb_p  = (_Float16*)carve(8192ull * 2);
  _Float16* mlp_p  = (_Float16*)carve(8192ull * 2);
  _Float16* hout_p = (_Float16*)carve(16384ull * 2);

  adj_kernel<<<NB * NN, NN, 0, stream>>>(pos, mask, adjw);

  // h = atom @ emb_w + emb_b
  pack_kernel<<<64, 256, 0, stream>>>(emb_w, 128, 64, 128, 2, 8, emb_p);
  gemm16<0, false><<<(256 * 8 + 7) / 8, 256, 0, stream>>>(
      atom, 64, 256, 2, emb_p, emb_b, 128, hbuf, NH, 8, nullptr, 0);

  for (int l = 0; l < 3; ++l) {
    const float* e_w1 = (const float*)d_in[9 + l * 15 + 0];
    const float* e_b1 = (const float*)d_in[9 + l * 15 + 1];
    const float* e_w2 = (const float*)d_in[9 + l * 15 + 2];
    const float* e_b2 = (const float*)d_in[9 + l * 15 + 3];
    const float* g_w  = (const float*)d_in[9 + l * 15 + 4];
    const float* g_b  = (const float*)d_in[9 + l * 15 + 5];
    const float* c_w1 = (const float*)d_in[9 + l * 15 + 6];
    const float* c_b1 = (const float*)d_in[9 + l * 15 + 7];
    const float* c_w2 = (const float*)d_in[9 + l * 15 + 8];
    const float* c_b2 = (const float*)d_in[9 + l * 15 + 9];
    const float* n_w1 = (const float*)d_in[9 + l * 15 + 10];
    const float* n_b1 = (const float*)d_in[9 + l * 15 + 11];
    const float* n_w2 = (const float*)d_in[9 + l * 15 + 12];
    const float* n_b2 = (const float*)d_in[9 + l * 15 + 13];
    const float* cscl = (const float*)d_in[9 + l * 15 + 14];

    pack_kernel<<<256, 256, 0, stream>>>(e_w1, 514, 128, 514, 4, 34, w1a_p);
    pack_kernel<<<256, 256, 0, stream>>>(e_w1 + 128 * 514, 514, 128, 514, 4, 34, w1b_p);
    pack_kernel<<<256, 256, 0, stream>>>(e_w2, 128, 514, 128, 17, 8, w2_p);
    pack_kernel<<<256, 256, 0, stream>>>(c_w1, 512, 128, 512, 4, 32, c1_p);
    pack_kernel<<<256, 256, 0, stream>>>(n_w1, 256, 256, 256, 8, 16, n1_p);
    pack_kernel<<<256, 256, 0, stream>>>(n_w2, 128, 256, 128, 8, 8, n2_p);

    // Hi = h @ W1a + b1 (544 padded), Hj = h @ W1b
    gemm16<0, false><<<(256 * 34 + 7) / 8, 256, 0, stream>>>(
        hbuf, NH, 256, 4, w1a_p, e_b1, 514, HiHj, LDHIHJ, 34, nullptr, 0);
    gemm16<0, false><<<(256 * 34 + 7) / 8, 256, 0, stream>>>(
        hbuf, NH, 256, 4, w1b_p, nullptr, 0, HiHj + HJOFF, LDHIHJ, 34, nullptr, 0);

    const float* xin = (l == 0) ? pos : ((l == 1) ? xA : xB);
    float* xout = (l == 0) ? xA : ((l == 1) ? xB : xA);
    edge_kernel<<<NB * 4, 128, 0, stream>>>(HiHj, xin, adjw, hbuf, w2_p, e_b2,
                                            g_w, g_b, c1_p, c_b1, c_w2, c_b2,
                                            e_w1, cscl, xout, hcat);

    // node MLP: nm = silu([h|m_i] @ n_w1 + b1); h += nm @ n_w2 + b2
    gemm16<1, false><<<(256 * 16 + 7) / 8, 256, 0, stream>>>(
        hcat, 256, 256, 8, n1_p, n_b1, 256, nmbuf, 256, 16, nullptr, 0);
    gemm16<0, true><<<(256 * 8 + 7) / 8, 256, 0, stream>>>(
        nmbuf, 256, 256, 8, n2_p, n_b2, 128, hbuf, NH, 8, hbuf, NH);
  }

  // output heads + masked scatter
  pack_kernel<<<128, 256, 0, stream>>>(how, 128, 128, 128, 4, 8, hout_p);
  pack_kernel<<<64, 256, 0, stream>>>(mlp_w, 128, 64, 128, 2, 8, mlp_p);
  gemm16<0, false><<<(256 * 8 + 7) / 8, 256, 0, stream>>>(
      hbuf, NH, 256, 4, hout_p, hob, 128, htmp, NH, 8, nullptr, 0);
  gemm16<0, false><<<(256 * 8 + 7) / 8, 256, 0, stream>>>(
      atom, 64, 256, 2, mlp_p, mlp_b, 128, hbase, NH, 8, nullptr, 0);
  finalize_kernel<<<2096, 256, 0, stream>>>(mask, htmp, hbase, xA, pos,
                                            (float*)d_out);
}
